// SSMHead_56977036149202
// MI455X (gfx1250) — matrix operand order repaired and checked by reading, not compiled, not run on hardware
//
#include <hip/hip_runtime.h>
#include <stdint.h>

#define B_SZ   8
#define T_LEN  2048
#define H_DIM  896
#define S_DIM  256
#define BT_ROWS (B_SZ * T_LEN)   // 16384

// ---------------------------------------------------------------------------
// WMMA types (CDNA5 / gfx1250, wave32)
// ---------------------------------------------------------------------------
typedef __attribute__((ext_vector_type(16))) __bf16 v16bf;
typedef __attribute__((ext_vector_type(8)))  float  v8f;

struct alignas(16) U4 { uint32_t x, y, z, w; };
struct U8 { U4 lo, hi; };   // 32 bytes == one 16x32 bf16 operand tile per lane

__device__ __forceinline__ v16bf tile_cast(const U8& t) {
  return __builtin_bit_cast(v16bf, t);
}

__device__ __forceinline__ v8f wmma_bf16(const U8& a, const U8& b, v8f c) {
  // D = A(16x32 bf16) x B(32x16 bf16) + C(16x16 f32)
  return __builtin_amdgcn_wmma_f32_16x16x32_bf16(
      /*neg_a=*/false, tile_cast(a), /*neg_b=*/false, tile_cast(b),
      /*c_mod=*/(short)0, c, /*reuse_a=*/false, /*reuse_b=*/false);
}

// f32 -> bf16 round-to-nearest-even
__device__ __forceinline__ uint32_t f2bf(float f) {
  uint32_t u = __builtin_bit_cast(uint32_t, f);
  return (u + 0x7FFFu + ((u >> 16) & 1u)) >> 16;
}
__device__ __forceinline__ uint32_t pack2bf(float lo, float hi) {
  return f2bf(lo) | (f2bf(hi) << 16);
}

// tanh via v_exp_f32 + v_rcp_f32 (bounded, clamp avoids inf/inf)
__device__ __forceinline__ float fast_tanh(float x) {
  x = fminf(8.0f, fmaxf(-8.0f, x));
  float e = exp2f(x * 2.8853900817779268f);   // e^(2x)
  return (e - 1.0f) * __builtin_amdgcn_rcpf(e + 1.0f);
}

// ---------------------------------------------------------------------------
// Precompute: tiled SGEMM (f32 x f32 -> bf16).  C[i][j] = sum_k A[i][k]*B[k][j]
// Used for M = Bm @ W_in  and  N = W_out @ Cm.  Off the critical path.
// ---------------------------------------------------------------------------
__global__ __launch_bounds__(256)
void k_sgemm_bf16(const float* __restrict__ A, const float* __restrict__ B,
                  uint16_t* __restrict__ C, int I, int K, int J) {
  __shared__ float As[16][16];
  __shared__ float Bs[16][17];
  const int tx = threadIdx.x & 15, ty = threadIdx.x >> 4;
  const int i = blockIdx.y * 16 + ty;
  const int j = blockIdx.x * 16 + tx;
  float acc = 0.0f;
  for (int kc = 0; kc < K; kc += 16) {
    As[ty][tx] = A[(size_t)i * K + kc + tx];
    Bs[ty][tx] = B[(size_t)(kc + ty) * J + j];
    __syncthreads();
#pragma unroll
    for (int kk = 0; kk < 16; ++kk) acc += As[ty][kk] * Bs[kk][tx];
    __syncthreads();
  }
  C[(size_t)i * J + j] = (uint16_t)f2bf(acc);
}

// bB[s] = sum_o Bm[s][o] * b_in[o]
__global__ void k_bias_proj(const float* __restrict__ Bm,
                            const float* __restrict__ b_in,
                            float* __restrict__ bB) {
  const int s = threadIdx.x;   // 256 threads
  float acc = 0.0f;
  for (int o = 0; o < H_DIM; ++o) acc += Bm[(size_t)s * H_DIM + o] * b_in[o];
  bB[s] = acc;
}

// ---------------------------------------------------------------------------
// GEMM1: xB[BT_ROWS][256] = cvt_bf16(x[BT_ROWS][896]) @ M^T + bB
//   M is [256][896] bf16 (row s = output col, contiguous in K) -> B operand
//   Block tile 128x128, 8 waves as 4(M) x 2(N), each wave 2x4 WMMA tiles.
//   LDS pitch 20 dwords/row: 16B-aligned b128 loads, conflict-free banks.
// ---------------------------------------------------------------------------
#define LDS_PITCH 20

__global__ __launch_bounds__(256)
void k_gemm_xB(const float* __restrict__ x, const uint16_t* __restrict__ Mw,
               const float* __restrict__ bB, float* __restrict__ xB) {
  __shared__ uint32_t As[128 * LDS_PITCH];
  __shared__ uint32_t Bs[128 * LDS_PITCH];
  const int tid  = threadIdx.x;
  const int row0 = blockIdx.y * 128;
  const int col0 = blockIdx.x * 128;
  const int lane = tid & 31, w = tid >> 5;
  const int m16  = lane & 15, half = lane >> 4;
  const int mw = w >> 1, nw = w & 1;

  const v8f zacc = {};
  v8f acc[2][4];
#pragma unroll
  for (int mt = 0; mt < 2; ++mt)
#pragma unroll
    for (int nt = 0; nt < 4; ++nt) acc[mt][nt] = zacc;

  const int fi = tid >> 1;           // 0..127: tile row to stage
  const int fk = (tid & 1) * 16;     // K sub-chunk

  for (int kc = 0; kc < H_DIM; kc += 32) {
    // stage A: 128x32 of x, f32 -> bf16 on the fly
    const float4* srcA = (const float4*)(x + (size_t)(row0 + fi) * H_DIM + kc + fk);
    float4 f0 = srcA[0], f1 = srcA[1], f2 = srcA[2], f3 = srcA[3];
    uint32_t* dA = &As[fi * LDS_PITCH + (fk >> 1)];
    dA[0] = pack2bf(f0.x, f0.y); dA[1] = pack2bf(f0.z, f0.w);
    dA[2] = pack2bf(f1.x, f1.y); dA[3] = pack2bf(f1.z, f1.w);
    dA[4] = pack2bf(f2.x, f2.y); dA[5] = pack2bf(f2.z, f2.w);
    dA[6] = pack2bf(f3.x, f3.y); dA[7] = pack2bf(f3.z, f3.w);
    // stage B (already bf16, stored N-major so it is B^T row-major)
    const U4* srcB = (const U4*)(Mw + (size_t)(col0 + fi) * H_DIM + kc + fk);
    U4* dB = (U4*)&Bs[fi * LDS_PITCH + (fk >> 1)];
    dB[0] = srcB[0]; dB[1] = srcB[1];
    __syncthreads();

    U8 a[2], b[4];
#pragma unroll
    for (int mt = 0; mt < 2; ++mt) {
      const int rm = mw * 32 + mt * 16 + m16;
      a[mt].lo = *(const U4*)&As[rm * LDS_PITCH + half * 4];        // K: half*8..+7
      a[mt].hi = *(const U4*)&As[rm * LDS_PITCH + 8 + half * 4];    // K: 16+half*8..
    }
#pragma unroll
    for (int nt = 0; nt < 4; ++nt) {
      const int cn = nw * 64 + nt * 16 + m16;
      b[nt].lo = *(const U4*)&Bs[cn * LDS_PITCH + half * 4];        // K: half*8..+7
      b[nt].hi = *(const U4*)&Bs[cn * LDS_PITCH + 8 + half * 4];    // K: 16+half*8..
    }
#pragma unroll
    for (int mt = 0; mt < 2; ++mt)
#pragma unroll
      for (int nt = 0; nt < 4; ++nt)
        acc[mt][nt] = wmma_bf16(a[mt], b[nt], acc[mt][nt]);
    __syncthreads();
  }

#pragma unroll
  for (int nt = 0; nt < 4; ++nt) {
    const int col = col0 + nw * 64 + nt * 16 + m16;
    const float bias = bB[col];
#pragma unroll
    for (int mt = 0; mt < 2; ++mt) {
      const int rbase = row0 + mw * 32 + mt * 16 + half * 8;
#pragma unroll
      for (int r = 0; r < 8; ++r)
        xB[(size_t)(rbase + r) * S_DIM + col] = acc[mt][nt][r] + bias;
    }
  }
}

// ---------------------------------------------------------------------------
// Sequential scan: h_t = tanh(h_{t-1} @ A^T + xB_t), all 8 batches in one
// padded 16x256 tile.  Single workgroup, 8 waves; wave w owns h columns
// [32w, 32w+32).  A^T stays resident in VGPRs as bf16 B-operands for all
// 2048 steps.  Ping-pong LDS H buffers -> one barrier per step.
// ---------------------------------------------------------------------------
#define HB_PITCH 132   // dwords per 256-col row: 16B aligned, spreads banks

__global__ __launch_bounds__(256)
void k_scan(const float* __restrict__ A, const float* __restrict__ xB,
            uint16_t* __restrict__ hseq) {
  __shared__ uint32_t Hb[2][16 * HB_PITCH];
  const int tid  = threadIdx.x;
  const int lane = tid & 31, w = tid >> 5;
  const int m16  = lane & 15, half = lane >> 4;
  const int n0   = w * 32;

  // h_0 = 0; pad rows 8..15 = 0 forever (never written -> never pollute)
  for (int q = tid; q < 2 * 16 * HB_PITCH; q += 256) ((uint32_t*)Hb)[q] = 0;

  // Resident B-operand: (A^T) element (k,n) = A[n0+n][k], bf16
  U8 Bt[2][8];
#pragma unroll
  for (int nt = 0; nt < 2; ++nt) {
    const float* ap = A + (size_t)(n0 + nt * 16 + m16) * S_DIM;
#pragma unroll
    for (int kt = 0; kt < 8; ++kt) {
      const int kb = kt * 32 + half * 8;
      uint32_t d[8];
#pragma unroll
      for (int v = 0; v < 8; ++v) {
        float2 f = *(const float2*)(ap + kb + (v >> 2) * 16 + 2 * (v & 3));
        d[v] = pack2bf(f.x, f.y);
      }
      Bt[nt][kt].lo = U4{d[0], d[1], d[2], d[3]};
      Bt[nt][kt].hi = U4{d[4], d[5], d[6], d[7]};
    }
  }
  __syncthreads();

  for (int t = 0; t < T_LEN; ++t) {
    const int cur = t & 1, nxt = cur ^ 1;

    // issue xB loads early (independent of recurrence) + prefetch next step
    float xb[2][8];
    if (half == 0) {
#pragma unroll
      for (int nt = 0; nt < 2; ++nt) {
        const int col = n0 + nt * 16 + m16;
#pragma unroll
        for (int r = 0; r < 8; ++r)
          xb[nt][r] = xB[((size_t)r * T_LEN + t) * S_DIM + col];
      }
      if (t + 1 < T_LEN)
        __builtin_prefetch(&xB[((size_t)(lane & 7) * T_LEN + (t + 1)) * S_DIM + n0], 0, 1);
    }

    // A-operand: current H (16x256 bf16) from LDS
    U8 a[8];
    const uint32_t* hb = &Hb[cur][m16 * HB_PITCH];
#pragma unroll
    for (int kt = 0; kt < 8; ++kt) {
      a[kt].lo = *(const U4*)&hb[kt * 16 + half * 4];
      a[kt].hi = *(const U4*)&hb[kt * 16 + 8 + half * 4];
    }

    v8f acc0 = {}, acc1 = {};
#pragma unroll
    for (int kt = 0; kt < 8; ++kt) acc0 = wmma_bf16(a[kt], Bt[0][kt], acc0);
#pragma unroll
    for (int kt = 0; kt < 8; ++kt) acc1 = wmma_bf16(a[kt], Bt[1][kt], acc1);

    // rows 0..7 (lanes 0..15 hold M = r): tanh, write next H + hseq
    if (half == 0) {
      uint16_t* hn = (uint16_t*)Hb[nxt];
      uint16_t* hg = hseq + (size_t)t * (8 * S_DIM);
#pragma unroll
      for (int nt = 0; nt < 2; ++nt) {
        const int col = n0 + nt * 16 + m16;
#pragma unroll
        for (int r = 0; r < 8; ++r) {
          const float v = (nt == 0 ? acc0[r] : acc1[r]) + xb[nt][r];
          const uint16_t u = (uint16_t)f2bf(fast_tanh(v));
          hn[r * (2 * HB_PITCH) + col] = u;
          hg[r * S_DIM + col] = u;
        }
      }
    }
    __syncthreads();
  }
}

// ---------------------------------------------------------------------------
// GEMM2: out[BT_ROWS][896] = hseq_rows(bf16) @ N^T + b_out
//   output row r = b*T + t  maps to hseq row t*8 + b;  N is [896][256] bf16.
// ---------------------------------------------------------------------------
__global__ __launch_bounds__(256)
void k_gemm_out(const uint16_t* __restrict__ hseq, const uint16_t* __restrict__ Nw,
                const float* __restrict__ b_out, float* __restrict__ out) {
  __shared__ uint32_t As[128 * LDS_PITCH];
  __shared__ uint32_t Bs[128 * LDS_PITCH];
  const int tid  = threadIdx.x;
  const int row0 = blockIdx.y * 128;
  const int col0 = blockIdx.x * 128;
  const int lane = tid & 31, w = tid >> 5;
  const int m16  = lane & 15, half = lane >> 4;
  const int mw = w >> 1, nw = w & 1;

  const v8f zacc = {};
  v8f acc[2][4];
#pragma unroll
  for (int mt = 0; mt < 2; ++mt)
#pragma unroll
    for (int nt = 0; nt < 4; ++nt) acc[mt][nt] = zacc;

  const int fi = tid >> 1;
  const int fk = (tid & 1) * 16;
  const int rg = row0 + fi;
  const int bb = rg >> 11;           // / T_LEN
  const int tt = rg & (T_LEN - 1);
  const uint16_t* arow = hseq + (size_t)(tt * 8 + bb) * S_DIM;

  for (int kc = 0; kc < S_DIM; kc += 32) {
    const U4* srcA = (const U4*)(arow + kc + fk);
    U4* dA = (U4*)&As[fi * LDS_PITCH + (fk >> 1)];
    dA[0] = srcA[0]; dA[1] = srcA[1];
    const U4* srcB = (const U4*)(Nw + (size_t)(col0 + fi) * S_DIM + kc + fk);
    U4* dB = (U4*)&Bs[fi * LDS_PITCH + (fk >> 1)];
    dB[0] = srcB[0]; dB[1] = srcB[1];
    __syncthreads();

    U8 a[2], b[4];
#pragma unroll
    for (int mt = 0; mt < 2; ++mt) {
      const int rm = mw * 32 + mt * 16 + m16;
      a[mt].lo = *(const U4*)&As[rm * LDS_PITCH + half * 4];
      a[mt].hi = *(const U4*)&As[rm * LDS_PITCH + 8 + half * 4];
    }
#pragma unroll
    for (int nt = 0; nt < 4; ++nt) {
      const int cn = nw * 64 + nt * 16 + m16;
      b[nt].lo = *(const U4*)&Bs[cn * LDS_PITCH + half * 4];
      b[nt].hi = *(const U4*)&Bs[cn * LDS_PITCH + 8 + half * 4];
    }
#pragma unroll
    for (int mt = 0; mt < 2; ++mt)
#pragma unroll
      for (int nt = 0; nt < 4; ++nt)
        acc[mt][nt] = wmma_bf16(a[mt], b[nt], acc[mt][nt]);
    __syncthreads();
  }

#pragma unroll
  for (int nt = 0; nt < 4; ++nt) {
    const int col = col0 + nw * 64 + nt * 16 + m16;
    const float bias = b_out[col];
#pragma unroll
    for (int mt = 0; mt < 2; ++mt) {
      const int rbase = row0 + mw * 32 + mt * 16 + half * 8;
#pragma unroll
      for (int r = 0; r < 8; ++r)
        out[(size_t)(rbase + r) * H_DIM + col] = acc[mt][nt][r] + bias;
    }
  }
}

// ---------------------------------------------------------------------------
// Launch: precompute folded weights, GEMM xB, sequential scan, output GEMM.
// Workspace layout (bytes from d_ws):
//   Mbf  [256*896] bf16   @ 0          (458752)
//   Nbf  [896*256] bf16   @ 458752     (458752)
//   bB   [256]     f32    @ 917504     (1024)
//   xB   [16384*256] f32  @ 918528     (16777216)
//   hseq [2048*8*256] bf16@ 17695744   (8388608)   total ~26 MB
// ---------------------------------------------------------------------------
extern "C" void kernel_launch(void* const* d_in, const int* in_sizes, int n_in,
                              void* d_out, int out_size, void* d_ws, size_t ws_size,
                              hipStream_t stream) {
  const float* x     = (const float*)d_in[0];
  const float* A     = (const float*)d_in[1];
  const float* Bm    = (const float*)d_in[2];
  const float* Cm    = (const float*)d_in[3];
  const float* W_in  = (const float*)d_in[4];
  const float* b_in  = (const float*)d_in[5];
  const float* W_out = (const float*)d_in[6];
  const float* b_out = (const float*)d_in[7];
  float* out = (float*)d_out;

  char* ws = (char*)d_ws;
  uint16_t* Mbf  = (uint16_t*)(ws);
  uint16_t* Nbf  = (uint16_t*)(ws + 458752);
  float*    bB   = (float*)   (ws + 917504);
  float*    xB   = (float*)   (ws + 918528);
  uint16_t* hseq = (uint16_t*)(ws + 17695744);

  // Folded weights: M = Bm @ W_in  [256][896];  N = W_out @ Cm  [896][256]
  k_sgemm_bf16<<<dim3(H_DIM / 16, S_DIM / 16), 256, 0, stream>>>(Bm, W_in, Mbf, S_DIM, H_DIM, H_DIM);
  k_sgemm_bf16<<<dim3(S_DIM / 16, H_DIM / 16), 256, 0, stream>>>(W_out, Cm, Nbf, H_DIM, H_DIM, S_DIM);
  k_bias_proj<<<1, 256, 0, stream>>>(Bm, b_in, bB);

  // xB = x @ M^T + bB
  k_gemm_xB<<<dim3(S_DIM / 128, BT_ROWS / 128), 256, 0, stream>>>(x, Mbf, bB, xB);

  // sequential recurrence (single WGP, A^T register-resident)
  k_scan<<<1, 256, 0, stream>>>(A, xB, hseq);

  // out = hseq @ N^T + b_out
  k_gemm_out<<<dim3(H_DIM / 128, BT_ROWS / 128), 256, 0, stream>>>(hseq, Nbf, b_out, out);
}
